// TrainableStreamingSNN_13365938225344
// MI455X (gfx1250) — matrix operand order repaired and checked by reading, not compiled, not run on hardware
//
#include <hip/hip_runtime.h>
#include <hip/hip_bf16.h>

#define TSEQ 512
#define NDIM 2048
#define EPSF 1e-8f
#define PI_F 3.14159265358979f

typedef __attribute__((ext_vector_type(16))) _Float16 v16h;
typedef __attribute__((ext_vector_type(8)))  _Float16 v8h;
typedef __attribute__((ext_vector_type(8)))  float    v8f;

__device__ __forceinline__ float sigmoidf_(float x) {
    return 1.0f / (1.0f + __expf(-x));
}

// torch/jax custom_sigmoid: guard = sigmoid(-5/x); x_safe = |x|>guard ? sign(x)*1e-8 : x;
// return sigmoid(-5/x_safe). (x==0 -> -5/0 = -inf -> sigmoid = 0, matching the limit.)
__device__ __forceinline__ float custom_sigmoid(float x) {
    float guard = sigmoidf_(-5.0f / x);
    float sgn = (x > 0.0f) ? 1.0f : ((x < 0.0f) ? -1.0f : 0.0f);
    float xs = (fabsf(x) > guard) ? sgn * 1e-8f : x;
    return sigmoidf_(-5.0f / xs);
}

// ---------------------------------------------------------------------------
// f32 -> f16 conversion (weights + x_seq)
// ---------------------------------------------------------------------------
__global__ __launch_bounds__(256)
void f32_to_f16_kernel(const float* __restrict__ src, _Float16* __restrict__ dst, int n) {
    int i = blockIdx.x * blockDim.x + threadIdx.x;
    if (i < n) dst[i] = (_Float16)src[i];
}

// ---------------------------------------------------------------------------
// Duty-cycle schedule: exact pos recurrence; emits sig_mod[t] = cs(mod*dcm_ss)
// and out_scale[t] = g_adapt*(1+mod).  Tiny; one thread.
// ---------------------------------------------------------------------------
__global__ void mod_kernel(const float* __restrict__ sc,
                           float* __restrict__ sig_mod,
                           float* __restrict__ out_scale, int Tn) {
    if (threadIdx.x != 0 || blockIdx.x != 0) return;
    float g_adapt  = sc[8];
    float cyc_len  = sc[9];
    float duty     = sc[10];
    float act_ms   = sc[11];
    float inact_ms = sc[12];
    float dcm_ss   = sc[13];
    float at = fminf(fmaxf(cyc_len * duty, 1.0f), cyc_len - 1.0f);
    float pos = 0.0f;
    for (int t = 0; t < Tn; ++t) {
        pos += 1.0f;
        if (pos >= cyc_len) pos = 0.0f;
        float is_active  = custom_sigmoid(10.0f * (at - pos));
        float progress   = pos / fmaxf(at, EPSF);
        float active_mod = act_ms + act_ms * sinf(progress * PI_F);
        float inact_prog = (pos - at) / fmaxf(cyc_len - at, EPSF);
        float inact_mod  = inact_ms * __expf(-3.0f * inact_prog);
        float mod = is_active * active_mod + (1.0f - is_active) * inact_mod;
        sig_mod[t]   = custom_sigmoid(mod * dcm_ss);
        out_scale[t] = g_adapt * (1.0f + mod);
    }
}

// ---------------------------------------------------------------------------
// Fragment helpers
// ---------------------------------------------------------------------------
// A fragment (16x32 f16): lanes 0-15 hold row M=lane, K {0..15};
// lanes 16-31 hold the same rows, K {16..31}.  p = &A[row][k0 + kA].
__device__ __forceinline__ v16h load_a_frag(const _Float16* __restrict__ p) {
    v8h lo = *(const v8h*)(p);
    v8h hi = *(const v8h*)(p + 8);
    v16h a;
#pragma unroll
    for (int i = 0; i < 8; ++i) { a[i] = lo[i]; a[8 + i] = hi[i]; }
    return a;
}

// Epilogue for one 16x16 D tile. D layout: VGPR r holds M = mBase + r,
// N = col; mBase already includes the +8 for lanes 16-31.
__device__ __forceinline__ void store_tile(v8f acc, int mBase, int col, int Nd,
                                           int mode, float a1s, float a2s,
                                           const float* __restrict__ bias,
                                           const float* __restrict__ rowscale,
                                           float* __restrict__ out_f,
                                           _Float16* __restrict__ out_h) {
    float bc = bias[col];
#pragma unroll
    for (int r = 0; r < 8; ++r) {
        int rr = mBase + r;
        float pre = acc[r] + bc;
        size_t idx = (size_t)rr * Nd + col;
        if (mode == 0) {
            out_h[idx] = (_Float16)custom_sigmoid(pre * a1s);
        } else if (mode == 1) {
            float h2 = custom_sigmoid(pre * a2s);
            out_f[idx] = h2 * custom_sigmoid(h2) * rowscale[rr];
        } else {
            out_f[idx] = custom_sigmoid(pre) * rowscale[rr];
        }
    }
}

// ---------------------------------------------------------------------------
// WMMA GEMM: C[T x Nd] = A[T x K] @ W[Nd x K]^T + bias, fused epilogue.
// 4 waves/block; each wave owns a 32(M) x 64(N) strip = 2 M-tiles x 4 N-tiles,
// 8 accumulators.  Per K=32 step: 4 A-loads + 8 B-loads feed 8 WMMAs
// (1.5 loads/WMMA), each A fragment reused 4x, each B fragment reused 2x.
// ---------------------------------------------------------------------------
__global__ __launch_bounds__(128)
void gemm_wmma_kernel(const _Float16* __restrict__ A,
                      const _Float16* __restrict__ W,
                      const float* __restrict__ bias,
                      const float* __restrict__ rowscale,
                      const float* __restrict__ scalars,
                      float* __restrict__ out_f,
                      _Float16* __restrict__ out_h,
                      int mode, int Nd, int Kd) {
    const int lane = threadIdx.x & 31;
    const int wave = threadIdx.x >> 5;
    const int m0 = blockIdx.y * 32;
    const int n0 = blockIdx.x * 256 + wave * 64;

    const int rowA = m0 + (lane & 15);        // M-tile 0 row; M-tile 1 = +16
    const int kA   = (lane >> 4) * 16;        // K sub-offset for A fragment
    const int colB = n0 + (lane & 15);        // N-tile 0 column (W row index)
    const int kB   = (lane >> 4) * 16;        // K sub-offset for B fragment

    v8f acc00 = {}, acc01 = {}, acc02 = {}, acc03 = {};
    v8f acc10 = {}, acc11 = {}, acc12 = {}, acc13 = {};

    const _Float16* __restrict__ arow0 = A + (size_t)rowA * Kd + kA;
    const _Float16* __restrict__ arow1 = arow0 + (size_t)16 * Kd;
    const _Float16* __restrict__ wrow  = W + (size_t)colB * Kd + kB;

    for (int k0 = 0; k0 < Kd; k0 += 32) {
        v16h a0 = load_a_frag(arow0 + k0);
        v16h a1 = load_a_frag(arow1 + k0);

        v16h b0 = *(const v16h*)(wrow + k0);
        v16h b1 = *(const v16h*)(wrow + k0 + (size_t)16 * Kd);
        v16h b2 = *(const v16h*)(wrow + k0 + (size_t)32 * Kd);
        v16h b3 = *(const v16h*)(wrow + k0 + (size_t)48 * Kd);

        acc00 = __builtin_amdgcn_wmma_f32_16x16x32_f16(false, a0, false, b0, (short)0, acc00, false, false);
        acc10 = __builtin_amdgcn_wmma_f32_16x16x32_f16(false, a1, false, b0, (short)0, acc10, false, false);
        acc01 = __builtin_amdgcn_wmma_f32_16x16x32_f16(false, a0, false, b1, (short)0, acc01, false, false);
        acc11 = __builtin_amdgcn_wmma_f32_16x16x32_f16(false, a1, false, b1, (short)0, acc11, false, false);
        acc02 = __builtin_amdgcn_wmma_f32_16x16x32_f16(false, a0, false, b2, (short)0, acc02, false, false);
        acc12 = __builtin_amdgcn_wmma_f32_16x16x32_f16(false, a1, false, b2, (short)0, acc12, false, false);
        acc03 = __builtin_amdgcn_wmma_f32_16x16x32_f16(false, a0, false, b3, (short)0, acc03, false, false);
        acc13 = __builtin_amdgcn_wmma_f32_16x16x32_f16(false, a1, false, b3, (short)0, acc13, false, false);
    }

    const float a1s = scalars[6];
    const float a2s = scalars[7];
    const int mBase0 = m0 + (lane >> 4) * 8;       // M-tile 0
    const int mBase1 = mBase0 + 16;                // M-tile 1

    store_tile(acc00, mBase0, colB +  0, Nd, mode, a1s, a2s, bias, rowscale, out_f, out_h);
    store_tile(acc01, mBase0, colB + 16, Nd, mode, a1s, a2s, bias, rowscale, out_f, out_h);
    store_tile(acc02, mBase0, colB + 32, Nd, mode, a1s, a2s, bias, rowscale, out_f, out_h);
    store_tile(acc03, mBase0, colB + 48, Nd, mode, a1s, a2s, bias, rowscale, out_f, out_h);
    store_tile(acc10, mBase1, colB +  0, Nd, mode, a1s, a2s, bias, rowscale, out_f, out_h);
    store_tile(acc11, mBase1, colB + 16, Nd, mode, a1s, a2s, bias, rowscale, out_f, out_h);
    store_tile(acc12, mBase1, colB + 32, Nd, mode, a1s, a2s, bias, rowscale, out_f, out_h);
    store_tile(acc13, mBase1, colB + 48, Nd, mode, a1s, a2s, bias, rowscale, out_f, out_h);
}

// ---------------------------------------------------------------------------
// LIF scan: per-neuron state is independent -> parallel over N, sequential in T.
// Produces spikes (half) feeding the final GEMM.
// ---------------------------------------------------------------------------
__global__ __launch_bounds__(256)
void lif_kernel(const float* __restrict__ xmod, const float* __restrict__ U,
                const float* __restrict__ sc, _Float16* __restrict__ spikes,
                int Tn, int Nn) {
    int n = blockIdx.x * blockDim.x + threadIdx.x;
    if (n >= Nn) return;
    float tau_mem = fminf(fmaxf(sc[0], 1.0f), 150.0f);
    float tau_syn = fminf(fmaxf(sc[1], 1.0f), 250.0f);
    float beta  = __expf(-1.0f / tau_mem);
    float alpha = __expf(-1.0f / tau_syn);
    float thresh  = fminf(fmaxf(sc[2], 0.1f), 5.0f);
    float reset_c = fminf(fmaxf(sc[3], -2.0f), 2.0f);
    float gain = sc[4], mem_nl = sc[5];

    float v = 0.0f, isyn = 0.0f;
    for (int t = 0; t < Tn; ++t) {
        size_t idx = (size_t)t * Nn + n;
        isyn = alpha * isyn + xmod[idx];
        v = beta * v + isyn * custom_sigmoid(v * mem_nl);
        float p = custom_sigmoid((v - thresh) * gain);
        float u = U[idx];
        float g = -__logf(-__logf(u + EPSF) + EPSF);
        float sp = sigmoidf_((__logf(p + EPSF) - __logf(1.0f - p + EPSF) + g) * 10.0f);
        float rs = custom_sigmoid(sp * 5.0f);
        v = v * (1.0f - rs) + reset_c * rs;
        spikes[idx] = (_Float16)sp;
    }
}

// ---------------------------------------------------------------------------
extern "C" void kernel_launch(void* const* d_in, const int* in_sizes, int n_in,
                              void* d_out, int out_size, void* d_ws, size_t ws_size,
                              hipStream_t stream) {
    const float* x  = (const float*)d_in[0];
    const float* U  = (const float*)d_in[1];
    const float* W1 = (const float*)d_in[2];
    const float* b1 = (const float*)d_in[3];
    const float* W2 = (const float*)d_in[4];
    const float* b2 = (const float*)d_in[5];
    const float* W3 = (const float*)d_in[6];
    const float* b3 = (const float*)d_in[7];
    const float* sc = (const float*)d_in[8];
    float* out = (float*)d_out;

    const int Tn = TSEQ, Nn = NDIM;
    const int TN = Tn * Nn, NN = Nn * Nn;

    // Workspace carve-up (256B aligned).
    char* ws = (char*)d_ws;
    size_t off = 0;
    auto carve = [&](size_t bytes) -> void* {
        void* p = ws + off;
        off = (off + bytes + 255) & ~(size_t)255;
        return p;
    };
    _Float16* Xh      = (_Float16*)carve((size_t)TN * 2);
    _Float16* W1h     = (_Float16*)carve((size_t)NN * 2);
    _Float16* W2h     = (_Float16*)carve((size_t)NN * 2);
    _Float16* W3h     = (_Float16*)carve((size_t)NN * 2);
    _Float16* H1h     = (_Float16*)carve((size_t)TN * 2);
    _Float16* Spikesh = (_Float16*)carve((size_t)TN * 2);
    float*    Xmod    = (float*)   carve((size_t)TN * 4);
    float*    sigmod  = (float*)   carve((size_t)Tn * 4);
    float*    outscale= (float*)   carve((size_t)Tn * 4);

    // Convert inputs to f16.
    f32_to_f16_kernel<<<(TN + 255) / 256, 256, 0, stream>>>(x,  Xh,  TN);
    f32_to_f16_kernel<<<(NN + 255) / 256, 256, 0, stream>>>(W1, W1h, NN);
    f32_to_f16_kernel<<<(NN + 255) / 256, 256, 0, stream>>>(W2, W2h, NN);
    f32_to_f16_kernel<<<(NN + 255) / 256, 256, 0, stream>>>(W3, W3h, NN);

    // Deterministic duty-cycle schedule.
    mod_kernel<<<1, 32, 0, stream>>>(sc, sigmod, outscale, Tn);

    dim3 gemm_grid(Nn / 256, Tn / 32);
    // Layer 1: H1 = cs((X @ W1^T + b1)*a1)
    gemm_wmma_kernel<<<gemm_grid, 128, 0, stream>>>(Xh, W1h, b1, nullptr, sc,
                                                    nullptr, H1h, 0, Nn, Nn);
    // Layer 2 + duty modulation: Xmod = h2*cs(h2)*sigmod[t]
    gemm_wmma_kernel<<<gemm_grid, 128, 0, stream>>>(H1h, W2h, b2, sigmod, sc,
                                                    Xmod, nullptr, 1, Nn, Nn);
    // Sequential LIF scan (elementwise in N).
    lif_kernel<<<(Nn + 255) / 256, 256, 0, stream>>>(Xmod, U, sc, Spikesh, Tn, Nn);
    // Layer 3 + output scaling: out = cs(S @ W3^T + b3) * g_adapt*(1+mod[t])
    gemm_wmma_kernel<<<gemm_grid, 128, 0, stream>>>(Spikesh, W3h, b3, outscale, sc,
                                                    out, nullptr, 2, Nn, Nn);
}
